// temporal_attention_33681133536110
// MI455X (gfx1250) — matrix operand order repaired and checked by reading, not compiled, not run on hardware
//
#include <hip/hip_runtime.h>

// Problem constants (fixed by reference): B=128, P=64, T=32, E=256, D=256
#define TT  32
#define EE  256
#define DD  256
#define K2  512               // 2*D for final linear
#define LDA  264              // f16 LDS row stride for enc_in tile (pad vs 256)
#define LDF  260              // f32 LDS row stride for enc tile
#define LDA2 520              // f16 LDS row stride for cat tile (pad vs 512)

typedef __attribute__((ext_vector_type(16))) _Float16 v16h;
typedef __attribute__((ext_vector_type(8)))  _Float16 v8h;
typedef __attribute__((ext_vector_type(8)))  float    v8f;
typedef __attribute__((ext_vector_type(4)))  float    v4f;

static __device__ __forceinline__ v16h load_afrag(const _Float16* p) {
    // A 16x32 f16 layout: this lane's 16 contiguous K elements
    v8h lo = *(const v8h*)(p);
    v8h hi = *(const v8h*)(p + 8);
    v16h a;
#pragma unroll
    for (int i = 0; i < 8; ++i) { a[i] = lo[i]; a[i + 8] = hi[i]; }
    return a;
}

// ---------------- Kernel 0: cast weights f32 -> f16 into workspace ----------------
__global__ __launch_bounds__(256) void k0_cast_weights(
    const float* __restrict__ We, const float* __restrict__ Wl,
    _Float16* __restrict__ We_h, _Float16* __restrict__ Wl_h)
{
    int i = blockIdx.x * 256 + threadIdx.x;            // 0 .. 131071
    if (i < DD * EE)  We_h[i] = (_Float16)We[i];
    if (i < DD * K2)  Wl_h[i] = (_Float16)Wl[i];
}

// ---------------- Kernel 1: per-(b,p) fused enc-GEMM + softmax + context ----------------
__global__ __launch_bounds__(256) void k1_attention(
    const float* __restrict__ dec_g,       // [BP, D]
    const float* __restrict__ enc_g,       // [BP, T, E]
    const float* __restrict__ mask_g,      // [BP, T]
    const _Float16* __restrict__ We_h,     // [D, E] f16
    float* __restrict__ out,               // [BP*D context staging | BP*T scores]
    int nBP)
{
    __shared__ _Float16 s_a[TT * LDA];     // enc_in tile as f16 (A matrix)
    __shared__ float    s_enc[TT * LDF];   // enc tile f32
    __shared__ float    s_dec[DD];
    __shared__ float    s_part[TT * 8];
    __shared__ float    s_score[TT];

    const int tid = threadIdx.x;
    const long bp = blockIdx.x;
    const float* enc_in = enc_g + bp * (long)(TT * EE);

    // ---- stage: enc_in [32,256] f32 -> f16 LDS (coalesced float4) ----
#pragma unroll
    for (int i = 0; i < 8; ++i) {
        int idx4 = tid + i * 256;              // 2048 float4 total
        int t  = idx4 >> 6;                    // 64 float4 per row
        int e4 = (idx4 & 63) << 2;
        v4f v = *(const v4f*)(enc_in + t * EE + e4);
        _Float16* dst = &s_a[t * LDA + e4];
        dst[0] = (_Float16)v[0]; dst[1] = (_Float16)v[1];
        dst[2] = (_Float16)v[2]; dst[3] = (_Float16)v[3];
    }
    s_dec[tid] = dec_g[bp * DD + tid];
    __syncthreads();

    // ---- GEMM1: enc[t,d] = sum_e A[t,e] * W_enc[d,e]  (A f16 from LDS, B f16 from L2) ----
    const int wv   = tid >> 5;
    const int lane = tid & 31;
    const int lh   = lane & 15;
    const bool hiHalf = lane >= 16;
    const int n0base = wv * 32;                // each wave: n-tiles {32w, 32w+16}, m-tiles {0,16}

    v8f acc[2][2] = {};
#pragma unroll
    for (int k0 = 0; k0 < EE; k0 += 32) {
        const int ka = k0 + (hiHalf ? 16 : 0);
        v16h a0 = load_afrag(&s_a[(lh)      * LDA + ka]);
        v16h a1 = load_afrag(&s_a[(16 + lh) * LDA + ka]);
        const int kb = k0 + (hiHalf ? 16 : 0);
#pragma unroll
        for (int nn = 0; nn < 2; ++nn) {
            const int n = n0base + nn * 16 + lh;
            v16h bf = *(const v16h*)(We_h + (long)n * EE + kb);   // row of W_enc: contiguous K
            acc[0][nn] = __builtin_amdgcn_wmma_f32_16x16x32_f16(false, a0, false, bf,
                                                               (short)0, acc[0][nn], false, false);
            acc[1][nn] = __builtin_amdgcn_wmma_f32_16x16x32_f16(false, a1, false, bf,
                                                               (short)0, acc[1][nn], false, false);
        }
    }

    // ---- spill enc tile f32 to LDS (C/D layout: M = r + 8*hiHalf, N = lh) ----
#pragma unroll
    for (int mm = 0; mm < 2; ++mm)
#pragma unroll
        for (int nn = 0; nn < 2; ++nn)
#pragma unroll
            for (int r = 0; r < 8; ++r) {
                int m = mm * 16 + r + (hiHalf ? 8 : 0);
                int n = n0base + nn * 16 + lh;
                s_enc[m * LDF + n] = acc[mm][nn][r];
            }
    __syncthreads();

    // ---- scores: s[t] = (enc[t,:] . dec) * sqrt(D)  (divide by SCALING = 1/16) ----
    {
        int t = tid & 31, seg = tid >> 5;
        float p = 0.f;
#pragma unroll
        for (int j = 0; j < 32; ++j)
            p += s_enc[t * LDF + seg * 32 + j] * s_dec[seg * 32 + j];
        s_part[t * 8 + seg] = p;
    }
    __syncthreads();
    if (tid < 32) {                               // wave 0: softmax over T=32
        float s = 0.f;
#pragma unroll
        for (int seg = 0; seg < 8; ++seg) s += s_part[tid * 8 + seg];
        s *= 16.0f;
        float mx = s;
#pragma unroll
        for (int o = 16; o > 0; o >>= 1) mx = fmaxf(mx, __shfl_xor(mx, o, 32));
        float ex = __expf(s - mx);
        float sm = ex;
#pragma unroll
        for (int o = 16; o > 0; o >>= 1) sm += __shfl_xor(sm, o, 32);
        float sc = (ex / sm) * mask_g[bp * TT + tid];   // post-softmax mask
        s_score[tid] = sc;
        out[(long)nBP * DD + bp * TT + tid] = sc;       // score output
    }
    __syncthreads();

    // ---- context[d] = sum_t score[t] * enc[t,d]; park in out-region for kernel 2 ----
    float ctx = 0.f;
#pragma unroll
    for (int t = 0; t < TT; ++t) ctx += s_score[t] * s_enc[t * LDF + tid];
    out[bp * (long)DD + tid] = ctx;
}

// ---------------- Kernel 2: out = tanh([context|dec] @ W_lin^T + b), in-place over context ----------------
__global__ __launch_bounds__(256) void k2_final(
    const float* __restrict__ dec_g,       // [BP, D]
    const _Float16* __restrict__ Wl_h,     // [D, 2D] f16
    const float* __restrict__ b_lin,       // [D]
    float* __restrict__ out)               // context in, out overwritten
{
    __shared__ _Float16 s_a[TT * LDA2];    // cat tile [32, 512] f16

    const int tid = threadIdx.x;
    const long row0 = (long)blockIdx.x * 32;

    // ---- stage cat rows: [0:256)=context (from out), [256:512)=decoder ----
#pragma unroll
    for (int i = 0; i < 8; ++i) {
        int idx4 = tid + i * 256;
        int t  = idx4 >> 6;
        int e4 = (idx4 & 63) << 2;
        v4f c = *(const v4f*)(out   + (row0 + t) * DD + e4);
        v4f d = *(const v4f*)(dec_g + (row0 + t) * DD + e4);
        _Float16* dc = &s_a[t * LDA2 + e4];
        dc[0] = (_Float16)c[0]; dc[1] = (_Float16)c[1];
        dc[2] = (_Float16)c[2]; dc[3] = (_Float16)c[3];
        _Float16* dd = &s_a[t * LDA2 + 256 + e4];
        dd[0] = (_Float16)d[0]; dd[1] = (_Float16)d[1];
        dd[2] = (_Float16)d[2]; dd[3] = (_Float16)d[3];
    }
    __syncthreads();   // all context reads complete before any out writes below

    const int wv   = tid >> 5;
    const int lane = tid & 31;
    const int lh   = lane & 15;
    const bool hiHalf = lane >= 16;
    const int n0base = wv * 32;

    v8f acc[2][2] = {};
#pragma unroll
    for (int k0 = 0; k0 < K2; k0 += 32) {
        const int ka = k0 + (hiHalf ? 16 : 0);
        v16h a0 = load_afrag(&s_a[(lh)      * LDA2 + ka]);
        v16h a1 = load_afrag(&s_a[(16 + lh) * LDA2 + ka]);
        const int kb = k0 + (hiHalf ? 16 : 0);
#pragma unroll
        for (int nn = 0; nn < 2; ++nn) {
            const int n = n0base + nn * 16 + lh;
            v16h bf = *(const v16h*)(Wl_h + (long)n * K2 + kb);   // row of W_lin: contiguous K
            acc[0][nn] = __builtin_amdgcn_wmma_f32_16x16x32_f16(false, a0, false, bf,
                                                               (short)0, acc[0][nn], false, false);
            acc[1][nn] = __builtin_amdgcn_wmma_f32_16x16x32_f16(false, a1, false, bf,
                                                               (short)0, acc[1][nn], false, false);
        }
    }

    // ---- epilogue: tanh(acc + b); overwrite context region ----
#pragma unroll
    for (int mm = 0; mm < 2; ++mm)
#pragma unroll
        for (int nn = 0; nn < 2; ++nn) {
            const int n = n0base + nn * 16 + lh;
            const float b = b_lin[n];
#pragma unroll
            for (int r = 0; r < 8; ++r) {
                int m = mm * 16 + r + (hiHalf ? 8 : 0);
                out[(row0 + m) * DD + n] = tanhf(acc[mm][nn][r] + b);
            }
        }
}

extern "C" void kernel_launch(void* const* d_in, const int* in_sizes, int n_in,
                              void* d_out, int out_size, void* d_ws, size_t ws_size,
                              hipStream_t stream) {
    const float* dec  = (const float*)d_in[0];   // [B,P,D]
    const float* enc  = (const float*)d_in[1];   // [B,P,1,T,E]
    const float* msk  = (const float*)d_in[2];   // [B,P,T]
    const float* We   = (const float*)d_in[3];   // [D,E]
    const float* Wl   = (const float*)d_in[4];   // [D,2D]
    const float* bl   = (const float*)d_in[5];   // [D]
    float* out = (float*)d_out;

    _Float16* We_h = (_Float16*)d_ws;            //  DD*EE f16 = 128 KB
    _Float16* Wl_h = We_h + DD * EE;             //  DD*K2 f16 = 256 KB

    const int nBP = in_sizes[0] / DD;            // 8192

    k0_cast_weights<<<(DD * K2 + 255) / 256, 256, 0, stream>>>(We, Wl, We_h, Wl_h);
    k1_attention<<<nBP, 256, 0, stream>>>(dec, enc, msk, We_h, out, nBP);
    k2_final<<<nBP / 32, 256, 0, stream>>>(dec, Wl_h, bl, out);
}
